// LSTM_Decoder_26199300505786
// MI455X (gfx1250) — matrix operand order repaired and checked by reading, not compiled, not run on hardware
//
#include <hip/hip_runtime.h>

// ---------------- types for CDNA5 WMMA ----------------
typedef __bf16  bf16;
typedef __bf16  bf16x8  __attribute__((ext_vector_type(8)));
typedef __bf16  v16bf   __attribute__((ext_vector_type(16)));
typedef float   v8f     __attribute__((ext_vector_type(8)));

// ---------------- problem constants ----------------
constexpr int Bn = 64, Sn = 512, Dn = 1024, An = 128, Cn = 64, TTn = 32, Tn = 40, NSn = 64;
constexpr float NEGV = -1000000000.0f;

// =====================================================================
// Prologue: f32 -> bf16 transpose-convert (weights to [N x K] layout)
// =====================================================================
__global__ void transpose_to_bf16(const float* __restrict__ src, bf16* __restrict__ dst,
                                  int K, int N, int dstLd, int colOff) {
  long total = (long)K * N;
  long stride = (long)gridDim.x * blockDim.x;
  for (long idx = (long)blockIdx.x * blockDim.x + threadIdx.x; idx < total; idx += stride) {
    long k = idx % K, n = idx / K;
    dst[n * (long)dstLd + colOff + k] = (bf16)src[k * (long)N + n];
  }
}

// =====================================================================
// Prologue: init state (c, h slot of xh), combined biases, zero output
// =====================================================================
__global__ void init_state(const float* __restrict__ h0, const float* __restrict__ c0,
                           const float* __restrict__ b_ih, const float* __restrict__ b_hh,
                           const float* __restrict__ b_act, const float* __restrict__ b_col,
                           const float* __restrict__ b_tab,
                           bf16* __restrict__ xh, float* __restrict__ cbuf,
                           float* __restrict__ biasG, float* __restrict__ biasP,
                           float* __restrict__ out) {
  int i = blockIdx.x * blockDim.x + threadIdx.x;
  if (i < Bn * Dn) {
    cbuf[i] = c0[i];
    int b = i / Dn, d = i % Dn;
    xh[(long)b * 4096 + 3072 + d] = (bf16)h0[i];
  }
  if (i < 4 * Dn) biasG[i] = b_ih[i] + b_hh[i];
  if (i < Dn) { biasP[i] = b_act[i]; biasP[Dn + i] = b_col[i]; biasP[2 * Dn + i] = b_tab[i]; }
  if (i < Bn) out[i] = 0.0f;
}

// =====================================================================
// WMMA GEMM: D(MxN,f32) = A(MxK,bf16) * Bt(NxK,bf16)^T + bias
// M == 64. One wave computes a 64x16 strip (4 accumulators).
// Block = 256 threads = 8 waves = 128 columns. grid.x = N/128.
// =====================================================================
__global__ void gemm_m64_bf16(const bf16* __restrict__ A, int lda,
                              const bf16* __restrict__ Bt, int ldb,
                              const float* __restrict__ bias,
                              float* __restrict__ Cf, bf16* __restrict__ Cb,
                              int ldc, int K) {
  const int wave = threadIdx.x >> 5;
  const int lane = threadIdx.x & 31;
  const int r = lane & 15;          // position within half-wave
  const int h = lane >> 4;          // half-wave select
  const int n0 = (blockIdx.x * 8 + wave) * 16;

  v8f acc[4] = { v8f{}, v8f{}, v8f{}, v8f{} };

  const bf16* brow = Bt + (long)(n0 + r) * ldb;
  const bf16* arow0 = A + (long)(r)      * lda;
  const bf16* arow1 = A + (long)(16 + r) * lda;
  const bf16* arow2 = A + (long)(32 + r) * lda;
  const bf16* arow3 = A + (long)(48 + r) * lda;

  for (int k0 = 0; k0 < K; k0 += 32) {
    // B fragment: elems 0..7 = K offsets 8h..8h+7; elems 8..15 = 16+8h..16+8h+7
    bf16x8 blo = *(const bf16x8*)(brow + k0 + 8 * h);
    bf16x8 bhi = *(const bf16x8*)(brow + k0 + 16 + 8 * h);
    v16bf bf = __builtin_shufflevector(blo, bhi, 0,1,2,3,4,5,6,7,8,9,10,11,12,13,14,15);

    // A fragments: elems 0..7 = K offsets 8h..8h+7; elems 8..15 = 16+8h..16+8h+7
    const bf16* ar[4] = { arow0, arow1, arow2, arow3 };
#pragma unroll
    for (int mt = 0; mt < 4; ++mt) {
      bf16x8 alo = *(const bf16x8*)(ar[mt] + k0 + 8 * h);
      bf16x8 ahi = *(const bf16x8*)(ar[mt] + k0 + 16 + 8 * h);
      v16bf af = __builtin_shufflevector(alo, ahi, 0,1,2,3,4,5,6,7,8,9,10,11,12,13,14,15);
      acc[mt] = __builtin_amdgcn_wmma_f32_16x16x32_bf16(
          false, af, false, bf, (short)0, acc[mt], false, false);
    }
  }

  const int n = n0 + r;
  const float bv = bias ? bias[n] : 0.0f;
#pragma unroll
  for (int mt = 0; mt < 4; ++mt) {
#pragma unroll
    for (int j = 0; j < 8; ++j) {
      int m = mt * 16 + 8 * h + j;      // C/D layout: VGPR j -> M=j (lanes 0-15), M=8+j (16-31)
      float v = acc[mt][j] + bv;
      if (Cf) Cf[(long)m * ldc + n] = v;
      if (Cb) Cb[(long)m * ldc + n] = (bf16)v;
    }
  }
}

// =====================================================================
// Per-step: build x = [prev_emb | node_emb | ... ] (bf16) into xh
// =====================================================================
__global__ void gather_x(const float* __restrict__ action_emb, const float* __restrict__ encoded_col,
                         const float* __restrict__ encoded_tab, const float* __restrict__ symbol_emb,
                         const int* __restrict__ cur_node_ids, const int* __restrict__ kind,
                         const int* __restrict__ gold_idx, bf16* __restrict__ xh, int t) {
  int b = blockIdx.x;
  int node = cur_node_ids[t * Bn + b];
  const float* pesrc = nullptr;
  if (t > 0) {
    int kd = kind[(t - 1) * Bn + b];
    int g  = gold_idx[(t - 1) * Bn + b];
    if (kd == 0)      pesrc = action_emb + (long)(((g % An) + An) % An) * Dn;
    else if (kd == 1) pesrc = encoded_col + ((long)b * Cn + ((g % Cn) + Cn) % Cn) * Dn;
    else              pesrc = encoded_tab + ((long)b * TTn + ((g % TTn) + TTn) % TTn) * Dn;
  }
  const float* nesrc = symbol_emb + (long)node * Dn;
  bf16* xb = xh + (long)b * 4096;
  for (int d = threadIdx.x; d < Dn; d += blockDim.x) {
    xb[d]        = (bf16)(pesrc ? pesrc[d] : 0.0f);
    xb[Dn + d]   = (bf16)nesrc[d];
  }
}

// =====================================================================
// Per-step: attention (scores, softmax, context) -> ctx slot of xh
// One block per batch row. Bandwidth-bound f32 scan of encoded_src.
// =====================================================================
__global__ void attention(const float* __restrict__ encoded_src, const int* __restrict__ src_mask,
                          const float* __restrict__ q, bf16* __restrict__ xh) {
  __shared__ float sc[Sn];
  __shared__ float red[256];
  int b = blockIdx.x;
  const float* src = encoded_src + (long)b * Sn * Dn;
  const float* qb  = q + (long)b * Dn;
  int wave = threadIdx.x >> 5, lane = threadIdx.x & 31;

  for (int s = wave; s < Sn; s += 8) {
    const float* row = src + (long)s * Dn;
    float acc = 0.0f;
    for (int d = lane; d < Dn; d += 32) acc += row[d] * qb[d];
    for (int off = 16; off; off >>= 1) acc += __shfl_xor(acc, off, 32);
    if (lane == 0) sc[s] = acc + (src_mask[b * Sn + s] == 1 ? NEGV : 0.0f);
  }
  __syncthreads();

  float m = -INFINITY;
  for (int s = threadIdx.x; s < Sn; s += 256) m = fmaxf(m, sc[s]);
  red[threadIdx.x] = m; __syncthreads();
  for (int st = 128; st; st >>= 1) {
    if (threadIdx.x < st) red[threadIdx.x] = fmaxf(red[threadIdx.x], red[threadIdx.x + st]);
    __syncthreads();
  }
  float mx = red[0]; __syncthreads();

  float sum = 0.0f;
  for (int s = threadIdx.x; s < Sn; s += 256) { float e = __expf(sc[s] - mx); sc[s] = e; sum += e; }
  red[threadIdx.x] = sum; __syncthreads();
  for (int st = 128; st; st >>= 1) {
    if (threadIdx.x < st) red[threadIdx.x] += red[threadIdx.x + st];
    __syncthreads();
  }
  float inv = 1.0f / red[0]; __syncthreads();
  for (int s = threadIdx.x; s < Sn; s += 256) sc[s] *= inv;
  __syncthreads();

  bf16* ctx = xh + (long)b * 4096 + 2048;
  for (int d = threadIdx.x; d < Dn; d += 256) {
    float acc = 0.0f;
    for (int s = 0; s < Sn; ++s) acc += sc[s] * src[(long)s * Dn + d];
    ctx[d] = (bf16)acc;
  }
}

// =====================================================================
// Per-step: LSTM pointwise; updates c (f32) and h slot of xh (bf16)
// =====================================================================
__global__ void lstm_cell(const float* __restrict__ gates, float* __restrict__ cbuf,
                          bf16* __restrict__ xh) {
  int i = blockIdx.x * blockDim.x + threadIdx.x;
  if (i >= Bn * Dn) return;
  int b = i / Dn, d = i % Dn;
  const float* g = gates + (long)b * 4096;
  float ig = g[d], fg = g[Dn + d], gg = g[2 * Dn + d], og = g[3 * Dn + d];
  float si = 1.0f / (1.0f + __expf(-ig));
  float sf = 1.0f / (1.0f + __expf(-fg));
  float so = 1.0f / (1.0f + __expf(-og));
  float c2 = sf * cbuf[i] + si * tanhf(gg);
  float h2 = so * tanhf(c2);
  cbuf[i] = c2;
  xh[(long)b * 4096 + 3072 + d] = (bf16)h2;
}

// =====================================================================
// Per-step: 224 logit dots per batch row, 3 log-softmaxes, loss gather
// =====================================================================
__global__ void loss_kernel(const bf16* __restrict__ projb,
                            const float* __restrict__ action_emb, const float* __restrict__ encoded_col,
                            const float* __restrict__ encoded_tab,
                            const int* __restrict__ am, const int* __restrict__ cm,
                            const int* __restrict__ tm,
                            const int* __restrict__ kind, const int* __restrict__ gold,
                            float* __restrict__ out, int t) {
  __shared__ float sc[An + Cn + TTn];   // 224 scores
  int b = blockIdx.x;
  int wave = threadIdx.x >> 5, lane = threadIdx.x & 31;
  const bf16* pb = projb + (long)b * 3072;

  for (int idx = wave; idx < An + Cn + TTn; idx += 8) {
    const float* vec; const bf16* x; float bias;
    if (idx < An)            { vec = action_emb + (long)idx * Dn;                     x = pb;            bias = (am[b * An + idx] == 1 ? NEGV : 0.0f); }
    else if (idx < An + Cn)  { int c = idx - An;  vec = encoded_col + ((long)b * Cn  + c) * Dn; x = pb + Dn;     bias = (cm[b * Cn + c] == 1 ? NEGV : 0.0f); }
    else                     { int tt = idx - An - Cn; vec = encoded_tab + ((long)b * TTn + tt) * Dn; x = pb + 2 * Dn; bias = (tm[b * TTn + tt] == 1 ? NEGV : 0.0f); }
    float acc = 0.0f;
    for (int d = lane; d < Dn; d += 32) acc += vec[d] * (float)x[d];
    for (int off = 16; off; off >>= 1) acc += __shfl_xor(acc, off, 32);
    if (lane == 0) sc[idx] = acc + bias;
  }
  __syncthreads();

  if (threadIdx.x == 0) {
    int kd = kind[t * Bn + b], g = gold[t * Bn + b];
    int base, len, gi;
    if (kd == 0)      { base = 0;       len = An;  gi = ((g % An)  + An)  % An;  }
    else if (kd == 1) { base = An;      len = Cn;  gi = ((g % Cn)  + Cn)  % Cn;  }
    else              { base = An + Cn; len = TTn; gi = ((g % TTn) + TTn) % TTn; }
    float mx = -INFINITY;
    for (int i = 0; i < len; ++i) mx = fmaxf(mx, sc[base + i]);
    float sum = 0.0f;
    for (int i = 0; i < len; ++i) sum += __expf(sc[base + i] - mx);
    float lp = sc[base + gi] - mx - __logf(sum);
    out[b] -= lp;
  }
}

// =====================================================================
// Host launcher
// =====================================================================
extern "C" void kernel_launch(void* const* d_in, const int* in_sizes, int n_in,
                              void* d_out, int out_size, void* d_ws, size_t ws_size,
                              hipStream_t stream) {
  const float* encoded_src = (const float*)d_in[0];
  const float* encoded_col = (const float*)d_in[1];
  const float* encoded_tab = (const float*)d_in[2];
  const float* h0          = (const float*)d_in[3];
  const float* c0          = (const float*)d_in[4];
  const float* action_emb  = (const float*)d_in[5];
  const float* symbol_emb  = (const float*)d_in[6];
  const float* W_tgt = (const float*)d_in[7];
  const float* b_tgt = (const float*)d_in[8];
  const float* W_act = (const float*)d_in[9];
  const float* b_act = (const float*)d_in[10];
  const float* W_col = (const float*)d_in[11];
  const float* b_col = (const float*)d_in[12];
  const float* W_tab = (const float*)d_in[13];
  const float* b_tab = (const float*)d_in[14];
  const float* W_out = (const float*)d_in[15];
  const float* b_out = (const float*)d_in[16];
  const float* W_ih  = (const float*)d_in[17];
  const float* b_ih  = (const float*)d_in[18];
  const float* W_hh  = (const float*)d_in[19];
  const float* b_hh  = (const float*)d_in[20];
  const int* src_mask    = (const int*)d_in[21];
  const int* col_mask    = (const int*)d_in[22];
  const int* tab_mask    = (const int*)d_in[23];
  const int* action_mask = (const int*)d_in[24];
  const int* cur_node_ids = (const int*)d_in[25];
  const int* kind         = (const int*)d_in[26];
  const int* gold_idx     = (const int*)d_in[27];
  float* out = (float*)d_out;

  // ---- workspace carve (256B aligned) ----
  char* p = (char*)d_ws;
  auto carve = [&](size_t bytes) -> void* {
    void* r = (void*)p;
    p += (bytes + 255) & ~(size_t)255;
    return r;
  };
  bf16*  wq_t    = (bf16*) carve((size_t)Dn * Dn * 2);        // W_tgt^T  [1024 x 1024]
  bf16*  wout_t  = (bf16*) carve((size_t)Dn * Dn * 2);        // W_out^T  [1024 x 1024]
  bf16*  wcat_t  = (bf16*) carve((size_t)4 * Dn * 4096 * 2);  // [W_ih;W_hh]^T [4096 x 4096]
  bf16*  wproj_t = (bf16*) carve((size_t)3 * Dn * Dn * 2);    // [W_act;W_col;W_tab]^T [3072 x 1024]
  float* biasG   = (float*)carve((size_t)4 * Dn * 4);
  float* biasP   = (float*)carve((size_t)3 * Dn * 4);
  bf16*  xh      = (bf16*) carve((size_t)Bn * 4096 * 2);      // [pe|ne|ctx|h] bf16
  float* cbuf    = (float*)carve((size_t)Bn * Dn * 4);
  float* qbuf    = (float*)carve((size_t)Bn * Dn * 4);
  float* gates   = (float*)carve((size_t)Bn * 4096 * 4);
  bf16*  outb    = (bf16*) carve((size_t)Bn * Dn * 2);
  bf16*  projb   = (bf16*) carve((size_t)Bn * 3072 * 2);

  // ---- prologue: weight transpose-convert (stays hot in 192MB L2) ----
  const int TB = 256, TG = 4096;
  transpose_to_bf16<<<TG, TB, 0, stream>>>(W_tgt, wq_t,    Dn,     Dn,     Dn,   0);
  transpose_to_bf16<<<TG, TB, 0, stream>>>(W_out, wout_t,  Dn,     Dn,     Dn,   0);
  transpose_to_bf16<<<TG, TB, 0, stream>>>(W_ih,  wcat_t,  3 * Dn, 4 * Dn, 4096, 0);
  transpose_to_bf16<<<TG, TB, 0, stream>>>(W_hh,  wcat_t,  Dn,     4 * Dn, 4096, 3072);
  transpose_to_bf16<<<TG, TB, 0, stream>>>(W_act, wproj_t,                Dn, Dn, Dn, 0);
  transpose_to_bf16<<<TG, TB, 0, stream>>>(W_col, wproj_t + (long)Dn * Dn,     Dn, Dn, Dn, 0);
  transpose_to_bf16<<<TG, TB, 0, stream>>>(W_tab, wproj_t + (long)2 * Dn * Dn, Dn, Dn, Dn, 0);
  init_state<<<(Bn * Dn + 255) / 256, 256, 0, stream>>>(h0, c0, b_ih, b_hh, b_act, b_col, b_tab,
                                                        xh, cbuf, biasG, biasP, out);

  // ---- sequential decode: 40 timesteps ----
  for (int t = 0; t < Tn; ++t) {
    gather_x<<<Bn, 256, 0, stream>>>(action_emb, encoded_col, encoded_tab, symbol_emb,
                                     cur_node_ids, kind, gold_idx, xh, t);
    // q = h @ W_tgt + b_tgt          (M=64, N=1024, K=1024)
    gemm_m64_bf16<<<Dn / 128, 256, 0, stream>>>(xh + 3072, 4096, wq_t, Dn, b_tgt,
                                                qbuf, nullptr, Dn, Dn);
    attention<<<Bn, 256, 0, stream>>>(encoded_src, src_mask, qbuf, xh);
    // gates = [pe|ne|ctx|h] @ [W_ih;W_hh] + (b_ih+b_hh)   (M=64, N=4096, K=4096)
    gemm_m64_bf16<<<4096 / 128, 256, 0, stream>>>(xh, 4096, wcat_t, 4096, biasG,
                                                  gates, nullptr, 4096, 4096);
    lstm_cell<<<(Bn * Dn + 255) / 256, 256, 0, stream>>>(gates, cbuf, xh);
    // out = h2 @ W_out + b_out       (M=64, N=1024, K=1024) -> bf16
    gemm_m64_bf16<<<Dn / 128, 256, 0, stream>>>(xh + 3072, 4096, wout_t, Dn, b_out,
                                                nullptr, outb, Dn, Dn);
    // proj = out @ [W_act|W_col|W_tab] + biases   (M=64, N=3072, K=1024) -> bf16
    gemm_m64_bf16<<<3072 / 128, 256, 0, stream>>>(outb, Dn, wproj_t, Dn, biasP,
                                                  nullptr, projb, 3072, Dn);
    loss_kernel<<<Bn, 256, 0, stream>>>(projb, action_emb, encoded_col, encoded_tab,
                                        action_mask, col_mask, tab_mask, kind, gold_idx, out, t);
  }
}
